// GroupQueryAttention_51616916963669
// MI455X (gfx1250) — matrix operand order repaired and checked by reading, not compiled, not run on hardware
//
#include <hip/hip_runtime.h>

// ---------------------------------------------------------------------------
// GQA layer for MI455X (gfx1250, wave32, WMMA 16x16x32 f16)
// B=1, S=2048, E=2048, H=32, G=8, D=64, GS=4, sliding window 128, sinks.
// ---------------------------------------------------------------------------

typedef __attribute__((ext_vector_type(16))) _Float16 v16h;
typedef __attribute__((ext_vector_type(8)))  _Float16 v8h;
typedef __attribute__((ext_vector_type(4)))  _Float16 v4h;
typedef __attribute__((ext_vector_type(8)))  float    v8f;
typedef __attribute__((ext_vector_type(4)))  float    v4f;

static constexpr int S_LEN = 2048;
static constexpr int EMB   = 2048;
static constexpr int NH    = 32;
static constexpr int NG    = 8;
static constexpr int HD    = 64;
static constexpr int SW    = 128;

static __device__ __forceinline__ v16h cat16(v8h lo, v8h hi) {
  v16h r;
#pragma unroll
  for (int i = 0; i < 8; ++i) { r[i] = lo[i]; r[8 + i] = hi[i]; }
  return r;
}

static __device__ __forceinline__ void wmma_f16(v16h a, v16h b, v8f& c) {
  c = __builtin_amdgcn_wmma_f32_16x16x32_f16(false, a, false, b, (short)0, c,
                                             false, false);
}

// ---------------------------------------------------------------------------
// One-shot fp32 -> f16 conversion (hoists all cvt out of the GEMM hot loops).
// ---------------------------------------------------------------------------
__global__ __launch_bounds__(256) void f32_to_f16(const float* __restrict__ in,
                                                  _Float16* __restrict__ out,
                                                  int n4) {
  const int i = blockIdx.x * blockDim.x + threadIdx.x;
  if (i >= n4) return;
  v4f f = *(const v4f*)(in + (size_t)i * 4);
  v4h h;
#pragma unroll
  for (int j = 0; j < 4; ++j) h[j] = (_Float16)f[j];
  *(v4h*)(out + (size_t)i * 4) = h;
}

// ---------------------------------------------------------------------------
// GEMM: C[M,N] = A[M,K] @ W[N,K]^T + bias[N].  One wave per 64x64 tile.
// A and W are f16 (pre-converted); bias/C fp32.
// WMMA fragment layouts used here (wave32; A/B in contiguous k order):
//   A 16x32 f16 : lane l -> row (l&15); elems 0..15 = K kb..kb+15,
//                 kb = k0 + 16*(l>>4)
//   B 32x16 f16 : lane l -> col (l&15); elems 0..15 = K ko..ko+15,
//                 ko = k0 + 16*(l>>4)
//   C 16x16 f32 : VGPR r, lane l -> row r + 8*(l>>4), col (l&15)
// Inner loop: 16 x global_load_b128 + 16 x v_wmma per 32-wide k-step.
// ---------------------------------------------------------------------------
__global__ __launch_bounds__(32) void gemm16(const _Float16* __restrict__ A,
                                             const _Float16* __restrict__ W,
                                             const float* __restrict__ bias,
                                             float* __restrict__ C,
                                             int M, int N, int K) {
  const int lane = threadIdx.x & 31;
  const int half = lane >> 4;
  const int lr   = lane & 15;
  const int tM = blockIdx.y * 64;
  const int tN = blockIdx.x * 64;

  v8f acc[4][4];
#pragma unroll
  for (int i = 0; i < 4; ++i)
#pragma unroll
    for (int j = 0; j < 4; ++j)
      acc[i][j] = (v8f){0.f,0.f,0.f,0.f,0.f,0.f,0.f,0.f};

  for (int k0 = 0; k0 < K; k0 += 32) {
    v16h af[4], bf[4];
    const int kb = k0 + 16 * half;
#pragma unroll
    for (int mi = 0; mi < 4; ++mi) {
      const _Float16* p = A + (size_t)(tM + mi * 16 + lr) * K + kb;
      af[mi] = cat16(*(const v8h*)p, *(const v8h*)(p + 8));
    }
    const int ko = k0 + 16 * half;
#pragma unroll
    for (int ni = 0; ni < 4; ++ni) {
      const _Float16* p = W + (size_t)(tN + ni * 16 + lr) * K + ko;
      bf[ni] = cat16(*(const v8h*)p, *(const v8h*)(p + 8));
    }
    if (k0 + 32 < K) {   // cover L2 latency for the next k-tile
      __builtin_prefetch(A + (size_t)(tM + lr) * K + kb + 32, 0, 3);
      __builtin_prefetch(W + (size_t)(tN + lr) * K + ko + 32, 0, 3);
    }
#pragma unroll
    for (int mi = 0; mi < 4; ++mi)
#pragma unroll
      for (int ni = 0; ni < 4; ++ni) wmma_f16(af[mi], bf[ni], acc[mi][ni]);
  }

#pragma unroll
  for (int ni = 0; ni < 4; ++ni) {
    const int col = tN + ni * 16 + lr;
    const float bc = bias[col];
#pragma unroll
    for (int mi = 0; mi < 4; ++mi)
#pragma unroll
      for (int r = 0; r < 8; ++r) {
        const int row = tM + mi * 16 + r + 8 * half;
        C[(size_t)row * N + col] = acc[mi][ni][r] + bc;
      }
  }
}

// ---------------------------------------------------------------------------
// RoPE (YaRN) + scatter into attention-friendly f16 layouts.
// grid = (S, H+2G), block = 32 (lane = rotary pair index j in [0,32)).
//   Q[f16][H][S][D]  (pre-scaled by 1/sqrt(D))
//   K[f16][G][S][D]
//   Vt[f16][G][D][S] (transposed so PV B-fragments are contiguous in S)
// ---------------------------------------------------------------------------
__global__ __launch_bounds__(32) void rope_scatter(
    const float* __restrict__ qkv, const int* __restrict__ pos_ids,
    _Float16* __restrict__ Q, _Float16* __restrict__ Kc,
    _Float16* __restrict__ Vt) {
  const int s    = blockIdx.x;
  const int slot = blockIdx.y;           // 0..47 : 32 Q heads, 8 K, 8 V
  const int j    = threadIdx.x & 31;     // rotary pair index

  const float x1 = qkv[(size_t)s * ((NH + 2 * NG) * HD) + slot * HD + j];
  const float x2 = qkv[(size_t)s * ((NH + 2 * NG) * HD) + slot * HD + j + 32];

  if (slot >= NH + NG) {                 // V: plain copy, transposed
    const int g = slot - NH - NG;
    Vt[((size_t)g * HD + j)      * S_LEN + s] = (_Float16)x1;
    Vt[((size_t)g * HD + j + 32) * S_LEN + s] = (_Float16)x2;
    return;
  }

  const float pos  = (float)pos_ids[s];
  const float freq = __powf(10000.0f, -(float)j * (1.0f / 32.0f));
  const float wl   = 6.28318530718f / freq;
  float t = (wl - 128.0f) * (1.0f / (4096.0f - 128.0f));
  t = fminf(fmaxf(t, 0.0f), 1.0f);
  const float eff  = freq * (1.0f - t) + 0.5f * freq * t;
  const float ang  = pos * eff * 1.06931471806f;   // 0.1*ln(2)+1
  float sn, cs;
  __sincosf(ang, &sn, &cs);

  float o1 = x1 * cs - x2 * sn;
  float o2 = x2 * cs + x1 * sn;

  if (slot < NH) {                        // Q head, fold in 1/sqrt(64)
    o1 *= 0.125f; o2 *= 0.125f;
    _Float16* qp = Q + ((size_t)slot * S_LEN + s) * HD;
    qp[j] = (_Float16)o1; qp[j + 32] = (_Float16)o2;
  } else {                                // K head
    const int g = slot - NH;
    _Float16* kp = Kc + ((size_t)g * S_LEN + s) * HD;
    kp[j] = (_Float16)o1; kp[j + 32] = (_Float16)o2;
  }
}

// ---------------------------------------------------------------------------
// Sliding-window attention with sinks. One wave per (16-query tile, head).
// Window 128 => at most 8 aligned 16-key tiles per query tile.
// Scores: Q(16x64) x K^T(64x16) = 2 WMMAs per key tile, staged in LDS f32.
// Softmax: row max/sum via __shfl_xor over the 16-lane group (matches the
// WMMA C layout: VGPR r / lane-half <-> row, lane&15 <-> column).
// P(16x128,f16 in LDS) x V^T -> ctx[S][H*D] f16.
// ---------------------------------------------------------------------------
__global__ __launch_bounds__(32) void attn16(const _Float16* __restrict__ Q,
                                             const _Float16* __restrict__ Kc,
                                             const _Float16* __restrict__ Vt,
                                             const float* __restrict__ sinks,
                                             _Float16* __restrict__ ctx) {
  const int lane = threadIdx.x & 31;
  const int half = lane >> 4;
  const int lr   = lane & 15;
  const int qt = blockIdx.x;
  const int h  = blockIdx.y;
  const int g  = h >> 2;                 // GS = 4
  const int q0 = qt * 16;

  __shared__ float    Sls[16 * 128];
  __shared__ _Float16 Pls[16 * 128];

  // Q A-fragments for K-dim 0..31 and 32..63, reused across all key tiles.
  v16h aq[2];
#pragma unroll
  for (int ks = 0; ks < 2; ++ks) {
    const int kb = ks * 32 + 16 * half;
    const _Float16* qp = Q + ((size_t)h * S_LEN + q0 + lr) * HD + kb;
    aq[ks] = cat16(*(const v8h*)qp, *(const v8h*)(qp + 8));
  }

  const int kt0    = (qt > 7) ? (qt - 7) : 0;
  const int ntiles = qt - kt0 + 1;       // 1..8

  float rmax[8];
#pragma unroll
  for (int r = 0; r < 8; ++r) rmax[r] = -1e30f;

  for (int kt = kt0; kt <= qt; ++kt) {
    v8f c = (v8f){0.f,0.f,0.f,0.f,0.f,0.f,0.f,0.f};
#pragma unroll
    for (int ks = 0; ks < 2; ++ks) {
      const int ko = ks * 32 + 16 * half;
      const _Float16* kp = Kc + ((size_t)g * S_LEN + kt * 16 + lr) * HD + ko;
      v16h bk = cat16(*(const v8h*)kp, *(const v8h*)(kp + 8));
      wmma_f16(aq[ks], bk, c);
    }
    const int key = kt * 16 + lr;
#pragma unroll
    for (int r = 0; r < 8; ++r) {
      const int qrow = q0 + r + 8 * half;
      const bool bad = (key > qrow) || (key <= qrow - SW);
      const float sc = bad ? -1e30f : c[r];
      Sls[(r + 8 * half) * 128 + (kt - kt0) * 16 + lr] = sc;
      rmax[r] = fmaxf(rmax[r], sc);
    }
  }

#pragma unroll
  for (int r = 0; r < 8; ++r) {
#pragma unroll
    for (int m = 1; m <= 8; m <<= 1)
      rmax[r] = fmaxf(rmax[r], __shfl_xor(rmax[r], m, 32));
  }
  const float snk = sinks[h];
#pragma unroll
  for (int r = 0; r < 8; ++r) rmax[r] = fmaxf(rmax[r], snk);

  __syncthreads();

  float rsum[8];
#pragma unroll
  for (int r = 0; r < 8; ++r) rsum[r] = 0.0f;
  for (int ti = 0; ti < ntiles; ++ti) {
#pragma unroll
    for (int r = 0; r < 8; ++r) {
      const int idx = (r + 8 * half) * 128 + ti * 16 + lr;
      const float p = __expf(Sls[idx] - rmax[r]);
      rsum[r] += p;
      Pls[idx] = (_Float16)p;
    }
  }
  if (ntiles & 1) {                      // pad to even #tiles for K=32 steps
#pragma unroll
    for (int r = 0; r < 8; ++r)
      Pls[(r + 8 * half) * 128 + ntiles * 16 + lr] = (_Float16)0.0f;
  }
#pragma unroll
  for (int r = 0; r < 8; ++r) {
#pragma unroll
    for (int m = 1; m <= 8; m <<= 1) rsum[r] += __shfl_xor(rsum[r], m, 32);
  }
  float rden[8];
#pragma unroll
  for (int r = 0; r < 8; ++r)
    rden[r] = 1.0f / (rsum[r] + __expf(snk - rmax[r]));

  __syncthreads();

  // ctx = P x V : output 16x64, K = 32 per step (2 key tiles).
  const int nks  = (ntiles + 1) >> 1;
  const int key0 = kt0 * 16;
  v8f acc[4];
#pragma unroll
  for (int d = 0; d < 4; ++d) acc[d] = (v8f){0.f,0.f,0.f,0.f,0.f,0.f,0.f,0.f};

  for (int ks = 0; ks < nks; ++ks) {
    const int kb = ks * 32 + 16 * half;
    const _Float16* pp = &Pls[lr * 128 + kb];
    v16h ap = cat16(*(const v8h*)pp, *(const v8h*)(pp + 8));
    const int ko = key0 + ks * 32 + 16 * half;
#pragma unroll
    for (int d = 0; d < 4; ++d) {
      const int col = d * 16 + lr;
      const _Float16* vp = Vt + ((size_t)g * HD + col) * S_LEN + ko;
      v16h bv = cat16(*(const v8h*)vp, *(const v8h*)(vp + 8));
      wmma_f16(ap, bv, acc[d]);
    }
  }

#pragma unroll
  for (int d = 0; d < 4; ++d)
#pragma unroll
    for (int r = 0; r < 8; ++r) {
      const int row = q0 + r + 8 * half;
      ctx[(size_t)row * (NH * HD) + h * HD + d * 16 + lr] =
          (_Float16)(acc[d][r] * rden[r]);
    }
}

// ---------------------------------------------------------------------------
// Launcher. Inputs: x, position_ids, attn_mask, Wqkv, bqkv, Wout, bout, sinks
// ---------------------------------------------------------------------------
extern "C" void kernel_launch(void* const* d_in, const int* in_sizes, int n_in,
                              void* d_out, int out_size, void* d_ws,
                              size_t ws_size, hipStream_t stream) {
  const float* x    = (const float*)d_in[0];
  const int*   pid  = (const int*)d_in[1];
  const float* Wqkv = (const float*)d_in[3];
  const float* bqkv = (const float*)d_in[4];
  const float* Wout = (const float*)d_in[5];
  const float* bout = (const float*)d_in[6];
  const float* sinks = (const float*)d_in[7];
  float* out = (float*)d_out;

  const int NQKV = (NH + 2 * NG) * HD;   // 3072

  // Workspace carve-up.
  uint8_t* ws = (uint8_t*)d_ws;
  size_t off = 0;
  float* qkv = (float*)(ws + off);
  off += (size_t)S_LEN * NQKV * sizeof(float);                     // 24 MB
  _Float16* Qh = (_Float16*)(ws + off);
  off += (size_t)NH * S_LEN * HD * sizeof(_Float16);               //  8 MB
  _Float16* Kh = (_Float16*)(ws + off);
  off += (size_t)NG * S_LEN * HD * sizeof(_Float16);               //  2 MB
  _Float16* Vt = (_Float16*)(ws + off);
  off += (size_t)NG * S_LEN * HD * sizeof(_Float16);               //  2 MB
  _Float16* ctx = (_Float16*)(ws + off);
  off += (size_t)S_LEN * NH * HD * sizeof(_Float16);               //  8 MB
  _Float16* xh = (_Float16*)(ws + off);
  off += (size_t)S_LEN * EMB * sizeof(_Float16);                   //  8 MB
  _Float16* Wqkvh = (_Float16*)(ws + off);
  off += (size_t)NQKV * EMB * sizeof(_Float16);                    // 12 MB
  _Float16* Wouth = (_Float16*)(ws + off);
  off += (size_t)EMB * EMB * sizeof(_Float16);                     //  8 MB

  // 0) One-shot fp32 -> f16 conversion of GEMM operands.
  {
    const int nx = S_LEN * EMB / 4, nw1 = NQKV * EMB / 4, nw2 = EMB * EMB / 4;
    f32_to_f16<<<(nx  + 255) / 256, 256, 0, stream>>>(x,    xh,    nx);
    f32_to_f16<<<(nw1 + 255) / 256, 256, 0, stream>>>(Wqkv, Wqkvh, nw1);
    f32_to_f16<<<(nw2 + 255) / 256, 256, 0, stream>>>(Wout, Wouth, nw2);
  }

  // 1) QKV projection (f16 WMMA, fp32 out + bias).
  gemm16<<<dim3(NQKV / 64, S_LEN / 64), 32, 0, stream>>>(
      xh, Wqkvh, bqkv, qkv, S_LEN, NQKV, EMB);

  // 2) YaRN RoPE + scatter to Q/K/V^T f16 layouts (Q pre-scaled by 1/8).
  rope_scatter<<<dim3(S_LEN, NH + 2 * NG), 32, 0, stream>>>(
      qkv, pid, Qh, Kh, Vt);

  // 3) Sliding-window attention with sinks -> ctx f16 [S][H*D].
  attn16<<<dim3(S_LEN / 16, NH), 32, 0, stream>>>(Qh, Kh, Vt, sinks, ctx);

  // 4) Output projection (f16 A/W, fp32 bias/out).
  gemm16<<<dim3(EMB / 64, S_LEN / 64), 32, 0, stream>>>(
      ctx, Wouth, bout, out, S_LEN, EMB, EMB);
}
